// MultiHeadAttention_47863115547133
// MI455X (gfx1250) — matrix operand order repaired and checked by reading, not compiled, not run on hardware
//
#include <hip/hip_runtime.h>

#define DEV __device__ __forceinline__

typedef __attribute__((ext_vector_type(16))) __bf16 v16bf;
typedef __attribute__((ext_vector_type(8)))  float  v8f;

union FragAB { v16bf v; unsigned int u[8]; };

DEV unsigned short f2bf(float f) {                 // f32 -> bf16 RNE
  unsigned int u = __float_as_uint(f);
  u += 0x7fffu + ((u >> 16) & 1u);
  return (unsigned short)(u >> 16);
}

DEV v8f wmma_bf16(const FragAB& a, const FragAB& b, v8f c) {
  // D(f32 16x16) = A(bf16 16x32) * B(bf16 32x16) + C
  return __builtin_amdgcn_wmma_f32_16x16x32_bf16(
      false, a.v, false, b.v, (short)0, c, false, false);
}

// ---- 16-lane xor-butterfly max via v_permlane16_b32 (pure VALU) -----------
DEV float plane_xor_max(float x, unsigned s0, unsigned s1) {
  int y = __builtin_amdgcn_permlane16(__float_as_int(x), __float_as_int(x),
                                      (int)s0, (int)s1, false, false);
  return fmaxf(x, __int_as_float(y));
}
DEV float rowmax16(float x) {  // max across the 16 lanes of each half-wave
  x = plane_xor_max(x, 0x67452301u, 0xEFCDAB89u);  // xor 1
  x = plane_xor_max(x, 0x54761032u, 0xDCFE98BAu);  // xor 2
  x = plane_xor_max(x, 0x32107654u, 0xBA98FEDCu);  // xor 4
  x = plane_xor_max(x, 0xFEDCBA98u, 0x76543210u);  // xor 8
  return x;
}

// ---- 4-element loaders: global (f32 or bf16 bits) -> bf16 shorts ----------
DEV void ld4(const float* p, unsigned short* d) {
  float4 f = *(const float4*)p;
  d[0] = f2bf(f.x); d[1] = f2bf(f.y); d[2] = f2bf(f.z); d[3] = f2bf(f.w);
}
DEV void ld4(const unsigned short* p, unsigned short* d) {
  uint2 q = *(const uint2*)p;
  d[0] = (unsigned short)(q.x & 0xffffu); d[1] = (unsigned short)(q.x >> 16);
  d[2] = (unsigned short)(q.y & 0xffffu); d[3] = (unsigned short)(q.y >> 16);
}

// ===========================================================================
// GEMM: out(M=4096,N=1024) = A(M,K=1024) * W(K,N) + bias, optional scale.
// 256 threads (8 waves); block tile 128x64; K-step 32; double-buffered LDS
// with global loads staged into registers one step ahead (1 barrier / step).
// Wave w -> rows 16w..16w+15, 4 WMMA accumulators across 64 cols.
// ===========================================================================
template <typename AT, bool OUT_QKV>
__global__ void __launch_bounds__(256)
gemm_bf16_kernel(const AT* __restrict__ A, const float* __restrict__ W,
                 const float* __restrict__ bias, void* __restrict__ outp,
                 float scale) {
  constexpr int Ndim = 1024, Kdim = 1024, KSTEPS = Kdim / 32;
  __shared__ unsigned int Alds_u[2][128 * 32 / 2];  // 128x32 bf16 row-major
  __shared__ unsigned int Wlds_u[2][64 * 32 / 2];   // 64(n) x 32(k) bf16

  const int tid  = threadIdx.x;
  const int wave = tid >> 5, lane = tid & 31;
  const int nb = blockIdx.x & 15;        // N/64
  const int mb = blockIdx.x >> 4;        // M/128
  const int Mbase = mb * 128, Nbase = nb * 64;
  const int rsub = tid >> 3, cg = tid & 7;

  unsigned short areg[16];   // staged A: 4 rows x 4 elems
  unsigned short wreg[8];    // staged W: 8 elems

  auto global_fetch = [&](int k0) {
#pragma unroll
    for (int p = 0; p < 4; p++) {
      int row = p * 32 + rsub;
      ld4(A + (size_t)(Mbase + row) * Kdim + k0 + cg * 4, areg + p * 4);
    }
#pragma unroll
    for (int i = 0; i < 8; i++) {
      int e = tid * 8 + i;
      int kk = e >> 6, n = e & 63;
      wreg[i] = f2bf(W[(size_t)(k0 + kk) * Ndim + Nbase + n]);
    }
  };
  auto lds_store = [&](int buf) {
    unsigned short* Al = (unsigned short*)Alds_u[buf];
    unsigned short* Wl = (unsigned short*)Wlds_u[buf];
#pragma unroll
    for (int p = 0; p < 4; p++) {
      int row = p * 32 + rsub;
#pragma unroll
      for (int j = 0; j < 4; j++) Al[row * 32 + cg * 4 + j] = areg[p * 4 + j];
    }
#pragma unroll
    for (int i = 0; i < 8; i++) {
      int e = tid * 8 + i;
      int kk = e >> 6, n = e & 63;
      Wl[n * 32 + kk] = wreg[i];
    }
  };

  v8f acc[4];
#pragma unroll
  for (int i = 0; i < 4; i++)
#pragma unroll
    for (int j = 0; j < 8; j++) acc[i][j] = 0.f;

  global_fetch(0);
  const int ml = wave * 16 + (lane & 15);
  const int kh = lane >> 4;

  for (int it = 0; it < KSTEPS; it++) {
    const int buf = it & 1;
    lds_store(buf);
    if (it + 1 < KSTEPS) global_fetch((it + 1) * 32);  // overlap w/ compute
    __syncthreads();  // double-buffered: single barrier per step

    const unsigned int* Au = Alds_u[buf];
    const unsigned int* Wu = Wlds_u[buf];
    FragAB af;
#pragma unroll
    for (int i = 0; i < 8; i++) {
      int kk = 16 * (i >> 2) + 8 * kh + 2 * (i & 3);
      af.u[i] = Au[(ml * 32 + kk) >> 1];
    }
    FragAB bfr[4];
#pragma unroll
    for (int nt = 0; nt < 4; nt++) {
      int nl = nt * 16 + (lane & 15);
#pragma unroll
      for (int i = 0; i < 8; i++) {
        int kk = 16 * (i >> 2) + 8 * kh + 2 * (i & 3);
        bfr[nt].u[i] = Wu[(nl * 32 + kk) >> 1];
      }
    }
#pragma unroll
    for (int nt = 0; nt < 4; nt++) acc[nt] = wmma_bf16(af, bfr[nt], acc[nt]);
  }

  // Epilogue: C layout row = r + 8*(lane/16), col = lane%16
  const int col = lane & 15, rh = lane >> 4;
#pragma unroll
  for (int nt = 0; nt < 4; nt++) {
    int n = Nbase + nt * 16 + col;
    float bv = bias[n];
#pragma unroll
    for (int r = 0; r < 8; r++) {
      int m = Mbase + wave * 16 + rh * 8 + r;
      float val = (acc[nt][r] + bv) * scale;
      if (OUT_QKV) {
        int bb = m >> 11, s = m & 2047;          // L = 2048
        int hh = n >> 6, dk = n & 63;            // DK = 64
        ((unsigned short*)outp)[((size_t)((bb * 16 + hh) * 2048 + s) << 6) + dk] =
            f2bf(val);
      } else {
        ((float*)outp)[(size_t)m * Ndim + n] = val;
      }
    }
  }
}

// ===========================================================================
// Flash attention: block = (b, h, 128 q rows); 8 waves x 16 q rows each.
// Q in registers (2 A-frags). Keys in chunks of 32, double-buffered K/V LDS
// tiles staged via registers (1 barrier / chunk):
//   S = Q*K^T (4 wmma) -> online softmax (permlane16 max butterfly)
//   -> P via per-wave LDS relayout -> ctx += P*V (4 wmma)
//   -> denominator via lsum += P*ones (1 wmma, alpha-rescaled like ctx).
// ===========================================================================
__global__ void __launch_bounds__(256)
attn_kernel(const unsigned short* __restrict__ q_ws,
            const unsigned short* __restrict__ k_ws,
            const unsigned short* __restrict__ v_ws,
            unsigned short* __restrict__ ctx_ws) {
  constexpr int Lseq = 2048, DKc = 64, NSTEP = Lseq / 32;
  __shared__ uint4 Klds4[2][256];                   // [key][dk] bf16 (32x64)
  __shared__ unsigned int Vlds_u[2][64 * 32 / 2];   // [dk][key] bf16 (64x32)
  __shared__ unsigned int Plds_u[8 * 16 * 32 / 2];  // per-wave 16x32 bf16 P

  const int tid = threadIdx.x, wave = tid >> 5, lane = tid & 31;
  const int qb = blockIdx.x & 15;
  const int h  = (blockIdx.x >> 4) & 15;
  const int b  = blockIdx.x >> 8;
  const size_t base = (size_t)(b * 16 + h) * Lseq * DKc;
  const int qrow0 = qb * 128 + wave * 16;
  const int ln = lane & 15, kg = lane >> 4;

  // Q fragments (A layout); 1/sqrt(DK) folded in at projection time.
  FragAB qf[2];
  {
    const unsigned int* qrow =
        (const unsigned int*)(q_ws + base + (size_t)(qrow0 + ln) * DKc);
#pragma unroll
    for (int f = 0; f < 2; f++)
#pragma unroll
      for (int i = 0; i < 8; i++) {
        int kd = 32 * f + 16 * (i >> 2) + 8 * kg + 2 * (i & 3);
        qf[f].u[i] = qrow[kd >> 1];
      }
  }

  FragAB onesf;  // B-frag of bf16 1.0 for denominator wmma
#pragma unroll
  for (int i = 0; i < 8; i++) onesf.u[i] = 0x3F803F80u;

  float mrun[8];
  v8f lsum, ctx[4];
#pragma unroll
  for (int r = 0; r < 8; r++) { mrun[r] = -1e30f; lsum[r] = 0.f; }
#pragma unroll
  for (int i = 0; i < 4; i++)
#pragma unroll
    for (int j = 0; j < 8; j++) ctx[i][j] = 0.f;

  uint4 kreg, vreg;  // staged K/V tile data
  {
    const uint4* ks = (const uint4*)(k_ws + base);
    const uint4* vs = (const uint4*)(v_ws + base);
    kreg = ks[tid]; vreg = vs[tid];
  }

  for (int it = 0; it < NSTEP; it++) {
    const int buf = it & 1;
    {  // store staged tile: K linear (b128), V transposed
      Klds4[buf][tid] = kreg;
      unsigned short* Vl = (unsigned short*)Vlds_u[buf];
      unsigned int wv[4] = {vreg.x, vreg.y, vreg.z, vreg.w};
      int e0 = tid * 8;
#pragma unroll
      for (int i = 0; i < 8; i++) {
        unsigned short val =
            (unsigned short)((wv[i >> 1] >> (16 * (i & 1))) & 0xffffu);
        int e = e0 + i, key = e >> 6, dk = e & 63;
        Vl[dk * 32 + key] = val;
      }
    }
    if (it + 1 < NSTEP) {  // fetch next tile while this one computes
      const uint4* ks = (const uint4*)(k_ws + base + (size_t)(it + 1) * 32 * DKc);
      const uint4* vs = (const uint4*)(v_ws + base + (size_t)(it + 1) * 32 * DKc);
      kreg = ks[tid]; vreg = vs[tid];
      if (it + 2 < NSTEP) {
        __builtin_prefetch(k_ws + base + (size_t)(it + 2) * 32 * DKc + tid * 8, 0, 3);
        __builtin_prefetch(v_ws + base + (size_t)(it + 2) * 32 * DKc + tid * 8, 0, 3);
      }
    }
    __syncthreads();  // double-buffered: single barrier per step

    // ---- S = Q * K^T : load all 4 K frags, then wmma burst ----
    const unsigned int* Ku = (const unsigned int*)Klds4[buf];
    FragAB kf[2][2];
#pragma unroll
    for (int t = 0; t < 2; t++)
#pragma unroll
      for (int f = 0; f < 2; f++)
#pragma unroll
        for (int i = 0; i < 8; i++) {
          int kd = 32 * f + 16 * (i >> 2) + 8 * kg + 2 * (i & 3);
          kf[t][f].u[i] = Ku[((16 * t + ln) * 64 + kd) >> 1];
        }
    v8f S[2];
#pragma unroll
    for (int t = 0; t < 2; t++) {
#pragma unroll
      for (int j = 0; j < 8; j++) S[t][j] = 0.f;
      S[t] = wmma_bf16(qf[0], kf[t][0], S[t]);
      S[t] = wmma_bf16(qf[1], kf[t][1], S[t]);
    }

    // ---- online softmax: row r of this lane = r + 8*(lane/16) ----
    float al[8];
#pragma unroll
    for (int r = 0; r < 8; r++) {
      float mv = rowmax16(fmaxf(S[0][r], S[1][r]));
      float mnew = fmaxf(mrun[r], mv);
      al[r] = __expf(mrun[r] - mnew);
      mrun[r] = mnew;
      S[0][r] = __expf(S[0][r] - mnew);
      S[1][r] = __expf(S[1][r] - mnew);
    }
#pragma unroll
    for (int nt = 0; nt < 4; nt++)
#pragma unroll
      for (int r = 0; r < 8; r++) ctx[nt][r] *= al[r];
#pragma unroll
    for (int r = 0; r < 8; r++) lsum[r] *= al[r];

    // ---- P: C-layout f32 -> per-wave LDS bf16 [m][k] -> A-layout ----
    unsigned short* Pl = ((unsigned short*)Plds_u) + wave * 16 * 32;
    {
      int rowh = kg * 8;
#pragma unroll
      for (int r = 0; r < 8; r++) {
        Pl[(rowh + r) * 32 + ln]      = f2bf(S[0][r]);
        Pl[(rowh + r) * 32 + 16 + ln] = f2bf(S[1][r]);
      }
    }
    asm volatile("s_wait_dscnt 0" ::: "memory");  // wave-local LDS RAW fence

    FragAB pf;
    {
      const unsigned int* Plu = (const unsigned int*)Pl;
#pragma unroll
      for (int i = 0; i < 8; i++) {
        int kk = 16 * (i >> 2) + 8 * kg + 2 * (i & 3);
        pf.u[i] = Plu[(ln * 32 + kk) >> 1];
      }
    }

    // ---- ctx += P*V (4 wmma), lsum += P*ones (1 wmma) ----
    const unsigned int* Vu = Vlds_u[buf];
    FragAB vf[4];
#pragma unroll
    for (int nt = 0; nt < 4; nt++) {
      int dk = nt * 16 + ln;
#pragma unroll
      for (int i = 0; i < 8; i++) {
        int key = 16 * (i >> 2) + 8 * kg + 2 * (i & 3);
        vf[nt].u[i] = Vu[(dk * 32 + key) >> 1];
      }
    }
    lsum = wmma_bf16(pf, onesf, lsum);
#pragma unroll
    for (int nt = 0; nt < 4; nt++) ctx[nt] = wmma_bf16(pf, vf[nt], ctx[nt]);
  }

  // ---- normalize, store ctx bf16 in (B, L, H*DK) row-major ----
  {
    float inv[8];
#pragma unroll
    for (int r = 0; r < 8; r++) inv[r] = 1.0f / lsum[r];
    int rh = lane >> 4;
#pragma unroll
    for (int nt = 0; nt < 4; nt++) {
#pragma unroll
      for (int r = 0; r < 8; r++) {
        int row = qrow0 + rh * 8 + r;
        float val = ctx[nt][r] * inv[r];
        int dk = nt * 16 + ln;
        ctx_ws[((size_t)(b * 2048 + row) << 10) + h * 64 + dk] = f2bf(val);
      }
    }
  }
}

// ===========================================================================
// Launch: Q/K/V projections -> flash attention -> output projection.
// Workspace: 4x 8MB bf16 tensors (Q,K,V in (B,H,L,DK); ctx in (B,L,H*DK)).
// ===========================================================================
extern "C" void kernel_launch(void* const* d_in, const int* in_sizes, int n_in,
                              void* d_out, int out_size, void* d_ws, size_t ws_size,
                              hipStream_t stream) {
  const float* query = (const float*)d_in[0];
  const float* key_  = (const float*)d_in[1];
  const float* value = (const float*)d_in[2];
  const float* Wq = (const float*)d_in[3];
  const float* bq = (const float*)d_in[4];
  const float* Wk = (const float*)d_in[5];
  const float* bk = (const float*)d_in[6];
  const float* Wv = (const float*)d_in[7];
  const float* bv = (const float*)d_in[8];
  const float* Wo = (const float*)d_in[9];
  const float* bo = (const float*)d_in[10];

  const size_t TEN = (size_t)4 * 1024 * 1024;     // elems per bf16 tensor (8 MB)
  unsigned short* q_ws   = (unsigned short*)d_ws;
  unsigned short* k_ws   = q_ws + TEN;
  unsigned short* v_ws   = k_ws + TEN;
  unsigned short* ctx_ws = v_ws + TEN;

  const int gemm_blocks = 512;   // (4096/128) * (1024/64)
  const int attn_blocks = 512;   // B * H * (L/128)
  const float scale = 0.125f;    // 1/sqrt(64), folded into Q

  gemm_bf16_kernel<float, true><<<gemm_blocks, 256, 0, stream>>>(
      query, Wq, bq, q_ws, scale);
  gemm_bf16_kernel<float, true><<<gemm_blocks, 256, 0, stream>>>(
      key_, Wk, bk, k_ws, 1.0f);
  gemm_bf16_kernel<float, true><<<gemm_blocks, 256, 0, stream>>>(
      value, Wv, bv, v_ws, 1.0f);

  attn_kernel<<<attn_blocks, 256, 0, stream>>>(q_ws, k_ws, v_ws, ctx_ws);

  gemm_bf16_kernel<unsigned short, false><<<gemm_blocks, 256, 0, stream>>>(
      ctx_ws, Wo, bo, d_out, 1.0f);
}
